// EncoderBlock_60455959658623
// MI455X (gfx1250) — matrix operand order repaired and checked by reading, not compiled, not run on hardware
//
#include <hip/hip_runtime.h>
#include <hip/hip_bf16.h>

typedef __bf16 bf16;
typedef __attribute__((ext_vector_type(16))) __bf16 v16bf;
typedef __attribute__((ext_vector_type(8)))  __bf16 v8bf;
typedef __attribute__((ext_vector_type(8)))  float  v8f;

#define HDIM  1024
#define SEQ   1024
#define BATCH 8
#define NH    16
#define DH    64

// ---------- WMMA helpers ----------
__device__ __forceinline__ v8f wmma_bf16(v16bf a, v16bf b, v8f c) {
  // D = A(16x32) * B(32x16) + C, f32 accumulate
  return __builtin_amdgcn_wmma_f32_16x16x32_bf16(false, a, false, b, (short)0, c,
                                                 false, false);
}

// Load a 16x32 (MxK) bf16 A-fragment. rowk0 = &A[row=lane&15][k0], hi = lane>>4.
// Lane<16: K = k0+0..15 (elems 0..15)
// Lane>=16: K = k0+16..31
__device__ __forceinline__ v16bf ld_afrag(const bf16* rowk0, int hi) {
  v8bf lo = *(const v8bf*)(rowk0 + hi * 16);
  v8bf hb = *(const v8bf*)(rowk0 + 8 + hi * 16);
  v16bf r;
#pragma unroll
  for (int i = 0; i < 8; ++i) { r[i] = lo[i]; r[i + 8] = hb[i]; }
  return r;
}

__device__ __forceinline__ float redmax16(float v) {
#pragma unroll
  for (int off = 1; off < 16; off <<= 1) v = fmaxf(v, __shfl_xor(v, off, 32));
  return v;
}
__device__ __forceinline__ float redsum16(float v) {
#pragma unroll
  for (int off = 1; off < 16; off <<= 1) v += __shfl_xor(v, off, 32);
  return v;
}

// ---------- 1) weights f32 -> bf16 ----------
__global__ void cvt_w_kernel(const float* __restrict__ q, const float* __restrict__ k,
                             const float* __restrict__ v, const float* __restrict__ w,
                             const float* __restrict__ u, bf16* __restrict__ dst) {
  const int widx = blockIdx.y;
  const size_t i = (size_t)blockIdx.x * 256 + threadIdx.x;
  const float* src = widx == 0 ? q : widx == 1 ? k : widx == 2 ? v : widx == 3 ? w : u;
  dst[(size_t)widx * HDIM * HDIM + i] = (bf16)src[i];
}

// ---------- 2) demodulation coefficients: dc[5][B][H] ----------
__global__ void demod_kernel(const float* __restrict__ qw, const float* __restrict__ kw,
                             const float* __restrict__ vw, const float* __restrict__ ww,
                             const float* __restrict__ uw, const float* __restrict__ styles,
                             float* __restrict__ dc) {
  const int t = blockIdx.x * 256 + threadIdx.x;
  if (t >= 5 * HDIM) return;
  const int widx = t >> 10, o = t & (HDIM - 1);
  const float* W =
      (widx == 0 ? qw : widx == 1 ? kw : widx == 2 ? vw : widx == 3 ? ww : uw) +
      (size_t)o * HDIM;
  const float* sb = styles + (widx >= 3 ? HDIM : 0);  // s1 for q/k/v, s2 for w/u
  float acc[BATCH] = {};
  for (int i = 0; i < HDIM; ++i) {
    const float w = W[i];
    const float w2 = w * w;
#pragma unroll
    for (int b = 0; b < BATCH; ++b) {
      const float sv = sb[(size_t)b * (2 * HDIM) + i];
      acc[b] += w2 * sv * sv;
    }
  }
#pragma unroll
  for (int b = 0; b < BATCH; ++b)
    dc[((size_t)widx * BATCH + b) * HDIM + o] = rsqrtf(acc[b] + 1e-8f);
}

// ---------- 3) style-scale + LayerNorm -> bf16 ----------
__global__ void ln_kernel(const float* __restrict__ x, const float* __restrict__ styles,
                          bf16* __restrict__ xn) {
  const int row = blockIdx.x;         // b*SEQ + s
  const int b = row >> 10;
  const int tid = threadIdx.x, wid = tid >> 5, lane = tid & 31;
  const float* xr = x + (size_t)row * HDIM;
  const float* s1 = styles + (size_t)b * (2 * HDIM);
  float v[4];
  float ls = 0.f, lq = 0.f;
#pragma unroll
  for (int i = 0; i < 4; ++i) {
    const int idx = tid + i * 256;
    v[i] = xr[idx] * s1[idx];
    ls += v[i];
    lq += v[i] * v[i];
  }
#pragma unroll
  for (int off = 16; off >= 1; off >>= 1) {
    ls += __shfl_xor(ls, off, 32);
    lq += __shfl_xor(lq, off, 32);
  }
  __shared__ float ssum[8], ssq[8];
  if (lane == 0) { ssum[wid] = ls; ssq[wid] = lq; }
  __syncthreads();
  float ts = 0.f, tq = 0.f;
#pragma unroll
  for (int w = 0; w < 8; ++w) { ts += ssum[w]; tq += ssq[w]; }
  const float mu = ts * (1.0f / HDIM);
  const float var = tq * (1.0f / HDIM) - mu * mu;
  const float rstd = rsqrtf(var + 1e-5f);
#pragma unroll
  for (int i = 0; i < 4; ++i) {
    const int idx = tid + i * 256;
    xn[(size_t)row * HDIM + idx] = (bf16)((v[i] - mu) * rstd);
  }
}

// ---------- 4) WMMA GEMM  Y = X(8192x1024) * W^T(1024x1024), fused epilogues ----------
// mode 0: Y *= dc  -> outa as [b,h,s,d]            (Q)
// mode 1: Y *= dc  -> outa as [b,h,s,d]            (K)
// mode 2: Y *= dc*s2 -> outa [b,s,o] (skip) AND outb [b,h,d,s] (V transposed)
// mode 3: Y *= dc  -> outf [b,s,o] f32 store       (W branch)
// mode 4: Y *= dc  -> outf [b,s,o] f32 accumulate  (U branch)
__global__ void gemm_kernel(const bf16* __restrict__ X, const bf16* __restrict__ W,
                            const float* __restrict__ dc, const float* __restrict__ styles,
                            float* __restrict__ outf, bf16* __restrict__ outa,
                            bf16* __restrict__ outb, int mode) {
  const int tid = threadIdx.x, wid = tid >> 5, lane = tid & 31;
  const int n16 = lane & 15, hi = lane >> 4;
  const int wm = wid & 3, wn = wid >> 2;
  const int mbase = blockIdx.x * 64 + wm * 16;   // M in [0,8192)
  const int nbase = blockIdx.y * 128 + wn * 64;  // N in [0,1024)
  const bf16* arow = X + (size_t)(mbase + n16) * HDIM;
  v8f acc[4] = {};
  for (int k0 = 0; k0 < HDIM; k0 += 32) {
    v16bf a = ld_afrag(arow + k0, hi);
    __builtin_prefetch(arow + k0 + 256, 0, 1);
#pragma unroll
    for (int c = 0; c < 4; ++c) {
      const bf16* wrow = W + (size_t)(nbase + c * 16 + n16) * HDIM + k0 + hi * 16;
      v16bf bb = *(const v16bf*)wrow;
      acc[c] = wmma_bf16(a, bb, acc[c]);
    }
  }
  const int b = mbase >> 10;
#pragma unroll
  for (int c = 0; c < 4; ++c) {
    const int n = nbase + c * 16 + n16;
    float mul = dc[(size_t)b * HDIM + n];
    if (mode == 2) mul *= styles[(size_t)b * (2 * HDIM) + HDIM + n];  // s2
#pragma unroll
    for (int e = 0; e < 8; ++e) {
      const int m = mbase + e + 8 * hi;   // D layout: row = e + 8*(lane>=16)
      const int s = m & (SEQ - 1);
      const float val = acc[c][e] * mul;
      if (mode <= 1) {
        const int hh = n >> 6, dd = n & 63;
        outa[(((size_t)b * NH + hh) * SEQ + s) * DH + dd] = (bf16)val;
      } else if (mode == 2) {
        outa[(size_t)m * HDIM + n] = (bf16)val;            // skip [b,s,o]
        const int hh = n >> 6, dd = n & 63;
        outb[(((size_t)b * NH + hh) * DH + dd) * SEQ + s] = (bf16)val;  // vT
      } else if (mode == 3) {
        outf[(size_t)m * HDIM + n] = val;
      } else {
        outf[(size_t)m * HDIM + n] += val;
      }
    }
  }
}

// ---------- 5) flash attention: per (b,h), wave = 16 query rows ----------
__global__ void attn_kernel(const bf16* __restrict__ q, const bf16* __restrict__ k,
                            const bf16* __restrict__ vT, bf16* __restrict__ o) {
  __shared__ bf16 plds[8][16][32];  // per-wave P-tile staging (D-layout -> A-layout)
  const int tid = threadIdx.x, wid = tid >> 5, lane = tid & 31;
  const int n16 = lane & 15, hi = lane >> 4;
  const int qb = blockIdx.x, h = blockIdx.y, b = blockIdx.z;
  const size_t head = (size_t)(b * NH + h);
  const bf16* qh = q + head * SEQ * DH;
  const bf16* kh = k + head * SEQ * DH;
  const bf16* vh = vT + head * DH * SEQ;
  const int m0 = qb * 128 + wid * 16;

  const bf16* qrow = qh + (size_t)(m0 + n16) * DH;
  v16bf qa0 = ld_afrag(qrow + 0, hi);   // Q K=0..31
  v16bf qa1 = ld_afrag(qrow + 32, hi);  // Q K=32..63

  v8f acc[4] = {};
  float mrun[8], lrun[8];
#pragma unroll
  for (int e = 0; e < 8; ++e) { mrun[e] = -1e30f; lrun[e] = 0.f; }
  const float scale = 0.125f;  // d^-0.5, d=64

  for (int kt = 0; kt < SEQ; kt += 32) {
    // scores: S = Q(16x64) * K_tile^T(64x32), as two 16x16 D tiles
    const bf16* krow0 = kh + (size_t)(kt + n16) * DH;
    const bf16* krow1 = kh + (size_t)(kt + 16 + n16) * DH;
    v8f s0 = {}, s1 = {};
    s0 = wmma_bf16(qa0, *(const v16bf*)(krow0 + hi * 16), s0);
    s0 = wmma_bf16(qa1, *(const v16bf*)(krow0 + 32 + hi * 16), s0);
    s1 = wmma_bf16(qa0, *(const v16bf*)(krow1 + hi * 16), s1);
    s1 = wmma_bf16(qa1, *(const v16bf*)(krow1 + 32 + hi * 16), s1);

    float alpha[8];
#pragma unroll
    for (int e = 0; e < 8; ++e) {
      const float a0 = s0[e] * scale;
      const float a1 = s1[e] * scale;
      const float mx = redmax16(fmaxf(a0, a1));
      const float mnew = fmaxf(mrun[e], mx);
      const float al = __expf(mrun[e] - mnew);
      const float p0 = __expf(a0 - mnew);
      const float p1 = __expf(a1 - mnew);
      lrun[e] = lrun[e] * al + redsum16(p0 + p1);
      mrun[e] = mnew;
      alpha[e] = al;
      const int row = e + 8 * hi;
      plds[wid][row][n16] = (bf16)p0;        // keys kt..kt+15
      plds[wid][row][n16 + 16] = (bf16)p1;   // keys kt+16..kt+31
    }
#pragma unroll
    for (int c = 0; c < 4; ++c)
#pragma unroll
      for (int e = 0; e < 8; ++e) acc[c][e] *= alpha[e];

    // reload P as A-fragment (per-wave DS ops are in-order; no barrier needed)
    v16bf pf = ld_afrag(&plds[wid][n16][0], hi);
#pragma unroll
    for (int c = 0; c < 4; ++c) {
      // B[k][n] = V[key=kt+k][dcol=c*16+n] = vT[dcol][kt+k]  -> contiguous row read
      v16bf bv = *(const v16bf*)(vh + (size_t)(c * 16 + n16) * SEQ + kt + hi * 16);
      acc[c] = wmma_bf16(pf, bv, acc[c]);
    }
  }

#pragma unroll
  for (int c = 0; c < 4; ++c)
#pragma unroll
    for (int e = 0; e < 8; ++e) {
      const int row = m0 + e + 8 * hi;
      const int col = h * DH + c * 16 + n16;
      const float val = acc[c][e] / lrun[e];
      o[((size_t)b * SEQ + row) * HDIM + col] = (bf16)val;
    }
}

// ---------- host launcher ----------
extern "C" void kernel_launch(void* const* d_in, const int* in_sizes, int n_in,
                              void* d_out, int out_size, void* d_ws, size_t ws_size,
                              hipStream_t stream) {
  const float* x = (const float*)d_in[0];
  const float* qw = (const float*)d_in[1];
  const float* kw = (const float*)d_in[2];
  const float* vw = (const float*)d_in[3];
  const float* ww = (const float*)d_in[4];
  const float* uw = (const float*)d_in[5];
  const float* styles = (const float*)d_in[6];
  float* out = (float*)d_out;

  const size_t HH = (size_t)HDIM * HDIM;
  const size_t ACT = (size_t)BATCH * SEQ * HDIM;  // 8M elems
  char* p = (char*)d_ws;
  bf16* Wbf = (bf16*)p;  p += 5 * HH * sizeof(bf16);
  bf16* xn = (bf16*)p;   p += ACT * sizeof(bf16);
  float* dc = (float*)p; p += (size_t)5 * BATCH * HDIM * sizeof(float);
  bf16* qb = (bf16*)p;   p += ACT * sizeof(bf16);
  bf16* kb = (bf16*)p;   p += ACT * sizeof(bf16);
  bf16* vT = (bf16*)p;   p += ACT * sizeof(bf16);
  bf16* skip = (bf16*)p; p += ACT * sizeof(bf16);
  bf16* ao = (bf16*)p;   p += ACT * sizeof(bf16);
  (void)in_sizes; (void)n_in; (void)out_size; (void)ws_size;

  cvt_w_kernel<<<dim3(HH / 256, 5), 256, 0, stream>>>(qw, kw, vw, ww, uw, Wbf);
  demod_kernel<<<(5 * HDIM + 255) / 256, 256, 0, stream>>>(qw, kw, vw, ww, uw, styles, dc);
  ln_kernel<<<BATCH * SEQ, 256, 0, stream>>>(x, styles, xn);

  const dim3 gg(8192 / 64, HDIM / 128);
  gemm_kernel<<<gg, 256, 0, stream>>>(xn, Wbf + 0 * HH, dc + 0 * BATCH * HDIM, styles,
                                      nullptr, qb, nullptr, 0);
  gemm_kernel<<<gg, 256, 0, stream>>>(xn, Wbf + 1 * HH, dc + 1 * BATCH * HDIM, styles,
                                      nullptr, kb, nullptr, 1);
  gemm_kernel<<<gg, 256, 0, stream>>>(xn, Wbf + 2 * HH, dc + 2 * BATCH * HDIM, styles,
                                      nullptr, skip, vT, 2);

  attn_kernel<<<dim3(SEQ / 128, NH, BATCH), 256, 0, stream>>>(qb, kb, vT, ao);

  gemm_kernel<<<gg, 256, 0, stream>>>(ao, Wbf + 3 * HH, dc + 3 * BATCH * HDIM, styles,
                                      out, nullptr, nullptr, 3);
  gemm_kernel<<<gg, 256, 0, stream>>>(skip, Wbf + 4 * HH, dc + 4 * BATCH * HDIM, styles,
                                      out, nullptr, nullptr, 4);
}
